// fGRUv2_47184510714633
// MI455X (gfx1250) — matrix operand order repaired and checked by reading, not compiled, not run on hardware
//
#include <hip/hip_runtime.h>
#include <hip/hip_bf16.h>
#include <math.h>

typedef __attribute__((ext_vector_type(16))) __bf16 v16bf;
typedef __attribute__((ext_vector_type(8)))  float  v8f;

#define HWD (128*128)

__device__ __forceinline__ unsigned short f2bf(float f) {
    unsigned int u = __float_as_uint(f);
    unsigned int r = u + 0x7FFFu + ((u >> 16) & 1u);
    return (unsigned short)(r >> 16);
}
__device__ __forceinline__ float bf2f(unsigned short h) {
    return __uint_as_float(((unsigned int)h) << 16);
}

// ---------------------------------------------------------------------------
// Weight prep: symmetrize conv weights, convert to bf16, swizzle into the
// A-fragment (16x32 bf16) per-lane layout so each lane reads 16 contiguous
// bf16 (32B). Layout: [ky*7+kx][ci_chunk(4)][co_tile(8)][lane(32)][elem(16)]
// A layout used here (contiguous k, same as B; not ISA table order): m = lane&15, half = lane>>4,
//   elem e -> krel = half*16 + e
// ---------------------------------------------------------------------------
__global__ void prep_weights(const float* __restrict__ Winh,
                             const float* __restrict__ Wexc,
                             const float* __restrict__ Wg,
                             const float* __restrict__ Wm,
                             unsigned short* __restrict__ wswI,
                             unsigned short* __restrict__ wswE,
                             unsigned short* __restrict__ wswG,
                             unsigned short* __restrict__ wswM) {
    const int NCONV = 49 * 4 * 8 * 32 * 16;  // 802816
    const int NG    = 12 * 8 * 32 * 16;      // 49152
    int gid = blockIdx.x * 256 + threadIdx.x;
    if (gid < 2 * NCONV) {
        const float* W = (gid < NCONV) ? Winh : Wexc;
        unsigned short* dst = (gid < NCONV) ? wswI : wswE;
        int idx = (gid < NCONV) ? gid : gid - NCONV;
        int e    = idx & 15;
        int lane = (idx >> 4) & 31;
        int cot  = (idx >> 9) & 7;
        int kc   = (idx >> 12) & 3;
        int kk   = idx >> 14;           // 0..48
        int ky = kk / 7, kx = kk % 7;
        int half = lane >> 4, m = lane & 15;
        int co = cot * 16 + m;
        int krel = half * 16 + e;
        int ci = kc * 32 + krel;
        float v = 0.5f * (W[((co * 128 + ci) * 7 + ky) * 7 + kx] +
                          W[((ci * 128 + co) * 7 + ky) * 7 + kx]);
        dst[idx] = f2bf(v);
    } else {
        int g = gid - 2 * NCONV;
        if (g < 2 * NG) {
            const float* W = (g < NG) ? Wg : Wm;
            unsigned short* dst = (g < NG) ? wswG : wswM;
            int idx = (g < NG) ? g : g - NG;
            int e    = idx & 15;
            int lane = (idx >> 4) & 31;
            int cot  = (idx >> 9) & 7;
            int kc   = idx >> 12;       // 0..11
            int half = lane >> 4, m = lane & 15;
            int co = cot * 16 + m;
            int krel = half * 16 + e;
            dst[idx] = f2bf(W[co * 384 + kc * 32 + krel]);
        }
    }
}

// ---------------------------------------------------------------------------
// gate_gemm: LN over 384 concat channels + 128x384 GEMM (WMMA bf16) + sigmoid.
// mode 0 (g1): sources [h_exc, h_inh, ff] NCHW f32; out = bf16(h_exc * g1)
// mode 1 (g2): sources [inhibited(NHWC bf16), h_exc, h_inh]; out = bf16(g2)
// WG = 256 thr (8 waves). Tile = 64 pixels x 128 out channels.
// ---------------------------------------------------------------------------
__global__ void gate_gemm(const float* __restrict__ hexc,
                          const float* __restrict__ hinh,
                          const float* __restrict__ ff,
                          const unsigned short* __restrict__ inh_t,
                          const unsigned short* __restrict__ wswA,
                          const float* __restrict__ bias,
                          unsigned short* __restrict__ out_t,
                          int mode) {
    __shared__ __attribute__((aligned(32))) unsigned short ldsin[64 * 384]; // 48KB
    __shared__ float red[8 * 64];
    __shared__ float meanA[64], rinvA[64];

    int tid = threadIdx.x;
    int p0 = blockIdx.x * 64;
    int b  = p0 >> 14;
    int y  = (p0 >> 7) & 127;
    int x0 = p0 & 127;

    // --- cooperative load into LDS (bf16), coalesced per source layout ---
    if (mode == 0) {
        for (int i = tid; i < 64 * 384; i += 256) {
            int c = i >> 6, px = i & 63;
            const float* s = (c < 128) ? hexc : ((c < 256) ? hinh : ff);
            int cc = c & 127;
            float v = s[((b * 128 + cc) * 128 + y) * 128 + x0 + px];
            ldsin[px * 384 + c] = f2bf(v);
        }
    } else {
        for (int i = tid; i < 64 * 128; i += 256) {       // inhibited, NHWC
            int px = i >> 7, c = i & 127;
            ldsin[px * 384 + c] = inh_t[(size_t)(p0 + px) * 128 + c];
        }
        for (int i = tid; i < 64 * 256; i += 256) {       // h_exc, h_inh NCHW
            int c = i >> 6, px = i & 63;
            const float* s = (c < 128) ? hexc : hinh;
            int cc = c & 127;
            float v = s[((b * 128 + cc) * 128 + y) * 128 + x0 + px];
            ldsin[px * 384 + 128 + c] = f2bf(v);
        }
    }
    __syncthreads();

    // --- LN stats over 384 channels per pixel ---
    {
        int px = tid & 63, grp = tid >> 6;
        float s = 0.f, ss = 0.f;
        for (int c = grp * 96; c < grp * 96 + 96; ++c) {
            float v = bf2f(ldsin[px * 384 + c]);
            s += v; ss += v * v;
        }
        red[grp * 64 + px] = s;
        red[256 + grp * 64 + px] = ss;
    }
    __syncthreads();
    if (tid < 64) {
        float s  = red[tid] + red[64 + tid] + red[128 + tid] + red[192 + tid];
        float ss = red[256 + tid] + red[320 + tid] + red[384 + tid] + red[448 + tid];
        float mean = s * (1.0f / 384.0f);
        float var  = ss * (1.0f / 384.0f) - mean * mean;
        meanA[tid] = mean;
        rinvA[tid] = rsqrtf(var + 1e-5f);
    }
    __syncthreads();
    {
        int px = tid & 63, grp = tid >> 6;
        float mean = meanA[px], rinv = rinvA[px];
        for (int c = grp * 96; c < grp * 96 + 96; ++c) {
            float v = bf2f(ldsin[px * 384 + c]);
            ldsin[px * 384 + c] = f2bf((v - mean) * rinv);
        }
    }
    __syncthreads();

    // --- WMMA GEMM: 128(co) x 64(px), K = 384 (12 chunks of 32) ---
    int wave = tid >> 5, lane = tid & 31;
    int half = lane >> 4, m16 = lane & 15;
    v8f z = {0.f, 0.f, 0.f, 0.f, 0.f, 0.f, 0.f, 0.f};
    v8f acc[4];
    acc[0] = z; acc[1] = z; acc[2] = z; acc[3] = z;

    const v16bf* A = (const v16bf*)wswA;
    for (int kc = 0; kc < 12; ++kc) {
        v16bf a = A[(kc * 8 + wave) * 32 + lane];
        int cb = kc * 32 + half * 16;   // B fragment: 16 contiguous K per lane
#pragma unroll
        for (int nt = 0; nt < 4; ++nt) {
            int px = nt * 16 + m16;
            v16bf bfrag = *(const v16bf*)&ldsin[px * 384 + cb];
            acc[nt] = __builtin_amdgcn_wmma_f32_16x16x32_bf16(
                false, a, false, bfrag, (short)0, acc[nt], false, false);
        }
    }
    __syncthreads();   // before reusing ldsin as staging

    // --- epilogue: sigmoid (+h_exc mult for mode 0), stage, coalesced store ---
    unsigned short* stage = ldsin;  // 64x128 bf16
#pragma unroll
    for (int nt = 0; nt < 4; ++nt) {
        int px = nt * 16 + m16;
#pragma unroll
        for (int r = 0; r < 8; ++r) {
            int co = wave * 16 + r + half * 8;
            float v = acc[nt][r] + bias[co];
            float g = 1.0f / (1.0f + __expf(-v));
            float outv = g;
            if (mode == 0) {
                float he = hexc[((b * 128 + co) * 128 + y) * 128 + x0 + px];
                outv = he * g;
            }
            stage[px * 128 + co] = f2bf(outv);
        }
    }
    __syncthreads();
    const uint4* sq = (const uint4*)stage;
    uint4* dq = (uint4*)(out_t + (size_t)p0 * 128);
    for (int i = tid; i < 1024; i += 256) dq[i] = sq[i];
}

// ---------------------------------------------------------------------------
// conv_wmma: implicit-GEMM 7x7 conv 128->128 + channel-LN + fused epilogue.
// WG = 256 thr (8 waves). Tile = 64 pixels (one row segment) x 128 co.
// stage 0: inhibited = relu(ff - relu((sig(alpha)*h_inh + mu) * LN(conv))),
//          written bf16 NHWC.
// stage 1: c2 = LN(conv); h_tilde = relu(k*(inh+c2)+relu(omega)*(inh*c2));
//          out = (1-g2)*h_exc + g2*h_tilde, written f32 NCHW.
// ---------------------------------------------------------------------------
__global__ void conv_wmma(const unsigned short* __restrict__ x_t,   // NHWC bf16
                          const unsigned short* __restrict__ wsw,   // swizzled A
                          const float* __restrict__ gamma,
                          const float* __restrict__ beta,
                          const float* __restrict__ ff,             // stage 0
                          const float* __restrict__ hinh,           // stage 0
                          const float* __restrict__ alpha,          // stage 0
                          const float* __restrict__ mu,             // stage 0
                          const unsigned short* __restrict__ inh_t, // stage 1
                          const unsigned short* __restrict__ g2_t,  // stage 1
                          const float* __restrict__ hexc,           // stage 1
                          const float* __restrict__ kappa,          // stage 1
                          const float* __restrict__ omega,          // stage 1
                          unsigned short* __restrict__ out_bf,      // stage 0
                          float* __restrict__ out_f32,              // stage 1
                          int stage) {
    __shared__ __attribute__((aligned(32))) unsigned short slab[70 * 128]; // 17.9KB
    __shared__ float outf[64 * 128];                                       // 32KB
    __shared__ float red[8 * 64];
    __shared__ float meanA[64], rinvA[64];
    __shared__ __attribute__((aligned(16))) unsigned short stg2[2 * 8192]; // 32KB

    int tid = threadIdx.x;
    int wg = blockIdx.x;
    int b  = wg >> 8;
    int y  = (wg >> 1) & 127;
    int x0 = (wg & 1) * 64;
    int p0 = (b * 128 + y) * 128 + x0;

    int wave = tid >> 5, lane = tid & 31;
    int half = lane >> 4, m16 = lane & 15;

    v8f z = {0.f, 0.f, 0.f, 0.f, 0.f, 0.f, 0.f, 0.f};
    v8f acc[4];
    acc[0] = z; acc[1] = z; acc[2] = z; acc[3] = z;

    const v16bf* A = (const v16bf*)wsw;

    for (int ky = 0; ky < 7; ++ky) {
        int ysrc = y + ky - 3;
        if (ysrc < 0 || ysrc > 127) continue;   // uniform across block
        __syncthreads();
        // stage input slab [px 0..69][ci 0..127] bf16 (x = x0-3+px)
        for (int i = tid; i < 70 * 128; i += 256) {
            int px = i >> 7, ci = i & 127;
            int xs = x0 - 3 + px;
            unsigned short v = 0;
            if (xs >= 0 && xs < 128)
                v = x_t[(size_t)((b * 128 + ysrc) * 128 + xs) * 128 + ci];
            slab[i] = v;
        }
        __syncthreads();
        for (int kx = 0; kx < 7; ++kx) {
            for (int kc = 0; kc < 4; ++kc) {
                v16bf a = A[(((ky * 7 + kx) * 4 + kc) * 8 + wave) * 32 + lane];
                int cb = kc * 32 + half * 16;
#pragma unroll
                for (int nt = 0; nt < 4; ++nt) {
                    int px = nt * 16 + m16 + kx;
                    v16bf bfrag = *(const v16bf*)&slab[px * 128 + cb];
                    acc[nt] = __builtin_amdgcn_wmma_f32_16x16x32_bf16(
                        false, a, false, bfrag, (short)0, acc[nt], false, false);
                }
            }
        }
    }
    __syncthreads();

    // spill accumulators to LDS for cross-wave channel-LN
#pragma unroll
    for (int nt = 0; nt < 4; ++nt) {
        int px = nt * 16 + m16;
#pragma unroll
        for (int r = 0; r < 8; ++r) {
            int co = wave * 16 + r + half * 8;
            outf[px * 128 + co] = acc[nt][r];
        }
    }
    __syncthreads();
    {
        int px = tid & 63, grp = tid >> 6;
        float s = 0.f, ss = 0.f;
        for (int c = grp * 32; c < grp * 32 + 32; ++c) {
            float v = outf[px * 128 + c];
            s += v; ss += v * v;
        }
        red[grp * 64 + px] = s;
        red[256 + grp * 64 + px] = ss;
    }
    __syncthreads();
    if (tid < 64) {
        float s  = red[tid] + red[64 + tid] + red[128 + tid] + red[192 + tid];
        float ss = red[256 + tid] + red[320 + tid] + red[384 + tid] + red[448 + tid];
        float mean = s * (1.0f / 128.0f);
        float var  = ss * (1.0f / 128.0f) - mean * mean;
        meanA[tid] = mean;
        rinvA[tid] = rsqrtf(var + 1e-5f);
    }
    __syncthreads();

    if (stage == 0) {
        unsigned short* stg = slab;   // conv done; reuse (8192 <= 8960)
        for (int i = tid; i < 8192; i += 256) {
            int co = i >> 6, px = i & 63;
            float c1 = (outf[px * 128 + co] - meanA[px]) * rinvA[px] * gamma[co] + beta[co];
            int gi = ((b * 128 + co) * 128 + y) * 128 + x0 + px;
            float f  = ff[gi];
            float hi = hinh[gi];
            float sA = 1.0f / (1.0f + __expf(-alpha[co]));
            float pre = (sA * hi + mu[co]) * c1;
            float inh = fmaxf(f - fmaxf(pre, 0.0f), 0.0f);
            stg[px * 128 + co] = f2bf(inh);
        }
        __syncthreads();
        const uint4* sq = (const uint4*)stg;
        uint4* dq = (uint4*)(out_bf + (size_t)p0 * 128);
        for (int i = tid; i < 1024; i += 256) dq[i] = sq[i];
    } else {
        unsigned short* inhS = stg2;
        unsigned short* g2S  = stg2 + 8192;
        {
            uint4* a4 = (uint4*)inhS;
            uint4* b4 = (uint4*)g2S;
            const uint4* ia = (const uint4*)(inh_t + (size_t)p0 * 128);
            const uint4* ib = (const uint4*)(g2_t + (size_t)p0 * 128);
            for (int i = tid; i < 1024; i += 256) { a4[i] = ia[i]; b4[i] = ib[i]; }
        }
        __syncthreads();
        for (int i = tid; i < 8192; i += 256) {
            int co = i >> 6, px = i & 63;
            float c2 = (outf[px * 128 + co] - meanA[px]) * rinvA[px] * gamma[co] + beta[co];
            float inh = bf2f(inhS[px * 128 + co]);
            float g2  = bf2f(g2S[px * 128 + co]);
            float k = kappa[co];
            float o = fmaxf(omega[co], 0.0f);
            float ht = fmaxf(k * (inh + c2) + o * (inh * c2), 0.0f);
            int gi = ((b * 128 + co) * 128 + y) * 128 + x0 + px;
            float he = hexc[gi];
            out_f32[gi] = (1.0f - g2) * he + g2 * ht;
        }
    }
}

// ---------------------------------------------------------------------------
extern "C" void kernel_launch(void* const* d_in, const int* in_sizes, int n_in,
                              void* d_out, int out_size, void* d_ws, size_t ws_size,
                              hipStream_t stream) {
    const float* ff    = (const float*)d_in[0];
    const float* hexc  = (const float*)d_in[1];
    const float* hinh  = (const float*)d_in[2];
    const float* Wg    = (const float*)d_in[3];
    const float* bg    = (const float*)d_in[4];
    const float* Wm    = (const float*)d_in[5];
    const float* bm    = (const float*)d_in[6];
    const float* Winh  = (const float*)d_in[7];
    const float* Wexc  = (const float*)d_in[8];
    const float* alpha = (const float*)d_in[9];
    const float* mu    = (const float*)d_in[10];
    const float* kappa = (const float*)d_in[11];
    const float* omega = (const float*)d_in[12];
    const float* c1g   = (const float*)d_in[13];
    const float* c1b   = (const float*)d_in[14];
    const float* c2g   = (const float*)d_in[15];
    const float* c2b   = (const float*)d_in[16];

    char* ws = (char*)d_ws;
    unsigned short* wswI = (unsigned short*)(ws);
    unsigned short* wswE = (unsigned short*)(ws + 1605632);
    unsigned short* wswG = (unsigned short*)(ws + 3211264);
    unsigned short* wswM = (unsigned short*)(ws + 3309568);
    unsigned short* x1t  = (unsigned short*)(ws + 3407872);
    unsigned short* inht = (unsigned short*)(ws + 3407872 + 33554432);
    unsigned short* g2t  = (unsigned short*)(ws + 3407872 + 2 * 33554432);

    prep_weights<<<6656, 256, 0, stream>>>(Winh, Wexc, Wg, Wm, wswI, wswE, wswG, wswM);

    // g1 + x1 = h_exc * g1  (bf16 NHWC)
    gate_gemm<<<2048, 256, 0, stream>>>(hexc, hinh, ff, nullptr, wswG, bg, x1t, 0);

    // conv1 + LN + inhibited (bf16 NHWC)
    conv_wmma<<<2048, 256, 0, stream>>>(x1t, wswI, c1g, c1b,
                                        ff, hinh, alpha, mu,
                                        nullptr, nullptr, nullptr, nullptr, nullptr,
                                        inht, nullptr, 0);

    // g2 (bf16 NHWC)
    gate_gemm<<<2048, 256, 0, stream>>>(hexc, hinh, ff, inht, wswM, bm, g2t, 1);

    // conv2 + LN + h_tilde + gate mix -> f32 NCHW output
    conv_wmma<<<2048, 256, 0, stream>>>(inht, wswE, c2g, c2b,
                                        nullptr, nullptr, nullptr, nullptr,
                                        inht, g2t, hexc, kappa, omega,
                                        nullptr, (float*)d_out, 1);
}
